// GNN_7215545057968
// MI455X (gfx1250) — matrix operand order repaired and checked by reading, not compiled, not run on hardware
//
#include <hip/hip_runtime.h>
#include <hip/hip_bf16.h>
#include <math.h>

#define GNN_N 50000
#define GNN_E 400000
#define GNN_H 128

typedef __bf16 bf16;
typedef __attribute__((ext_vector_type(16))) __bf16 v16bf;
typedef __attribute__((ext_vector_type(8)))  float  v8f;

__device__ __forceinline__ float silu_f(float x) { return x / (1.0f + __expf(-x)); }

__device__ __forceinline__ v8f wmma_bf16(v16bf a, v16bf b, v8f c) {
  // D = A(16x32 bf16) x B(32x16 bf16) + C(16x16 f32)
  return __builtin_amdgcn_wmma_f32_16x16x32_bf16(false, a, false, b, (short)0, c,
                                                 false, false);
}

// A fragment: rows [0,16) x cols [kb,kb+32) of row-major bf16 matrix A (leading dim lda).
// ISA layout: lane&15 = M; lane>>4 selects K-octet within each 16-wide half.
__device__ __forceinline__ v16bf load_a_frag(const bf16* A, int lda, int kb, int lane) {
  const int m = lane & 15;
  const int h = lane >> 4;
  v16bf a;
#pragma unroll
  for (int v = 0; v < 8; ++v) {
    const int k = kb + ((v & 4) << 2) + h * 8 + (v & 3) * 2;
    const bf16* p = A + m * lda + k;
    a[2 * v]     = p[0];
    a[2 * v + 1] = p[1];
  }
  return a;
}

// B fragment from nn.Linear weight W[out, K] (row-major): B(k, n) = W[n0+n][kb+k].
// ISA layout: lane&15 = N; lane>>4 selects K-octet within each 16-wide half.
__device__ __forceinline__ v16bf load_b_frag(const bf16* W, int ldw, int n0, int kb,
                                             int lane) {
  const int n = n0 + (lane & 15);
  const int h = lane >> 4;
  const bf16* p = W + (long)n * ldw + kb + h * 8;
  v16bf b;
#pragma unroll
  for (int v = 0; v < 16; ++v) b[v] = p[v + ((v & 8) ? 8 : 0)];
  return b;
}

template <int K>
__device__ __forceinline__ v8f gemm_tile(const bf16* A, int lda, const bf16* W, int ldw,
                                         int n0, int lane) {
  v8f c = {};
#pragma unroll
  for (int kb = 0; kb < K; kb += 32)
    c = wmma_bf16(load_a_frag(A, lda, kb, lane), load_b_frag(W, ldw, n0, kb, lane), c);
  return c;
}

// m2 GEMM (K=128) + silu + store messages bf16 + fp32 scatter-add into sums[recv].
__device__ __forceinline__ void msg_m2_scatter(const bf16* sH /*wave-local 16x128*/,
                                               const bf16* W2, const float* B2,
                                               const int* __restrict__ recv,
                                               bf16* __restrict__ hout,
                                               float* __restrict__ sums, long e0,
                                               int wave, int lane) {
  const int n = lane & 15, h = lane >> 4;
  long eRow[8];
  int rIdx[8];
#pragma unroll
  for (int r = 0; r < 8; ++r) {
    long e = e0 + wave * 16 + r + 8 * h;
    if (e >= GNN_E) e = GNN_E - 1;
    eRow[r] = e;
    rIdx[r] = recv[e];
  }
  for (int nt = 0; nt < 8; ++nt) {
    v8f c = gemm_tile<128>(sH, 128, W2, 128, nt * 16, lane);
    const float bias = B2[nt * 16 + n];
    const int col = nt * 16 + n;
#pragma unroll
    for (int r = 0; r < 8; ++r) {
      const float v = silu_f(c[r] + bias);
      hout[eRow[r] * GNN_H + col] = (bf16)v;
      atomicAdd(&sums[(long)rIdx[r] * GNN_H + col], v);
    }
  }
}

// Layers 2-4 edge MLP: feat = concat(x[send], x[recv], h_prev) [64,384] staged in LDS.
__global__ __launch_bounds__(128) void edge_mlp_kernel(
    const bf16* __restrict__ xbf, const int* __restrict__ send,
    const int* __restrict__ recv, const bf16* __restrict__ W1,
    const float* __restrict__ B1, const bf16* __restrict__ W2,
    const float* __restrict__ B2, bf16* __restrict__ hbuf /*in: prev msg, out: new*/,
    float* __restrict__ sums) {
  __shared__ bf16 sF[64 * 384];  // 48 KB
  __shared__ bf16 sH[64 * 128];  // 16 KB
  const int tid = threadIdx.x, lane = tid & 31, wave = tid >> 5;
  const long e0 = (long)blockIdx.x * 64;
  {
    const int row = tid >> 1, part = tid & 1;
    long e = e0 + row;
    if (e >= GNN_E) e = GNN_E - 1;
    const int s = send[e], rr = recv[e];
    const bf16* xs = xbf + (long)s * GNN_H;
    const bf16* xr = xbf + (long)rr * GNN_H;
    const bf16* hp = hbuf + e * GNN_H;
    const int base = part * 192;
#pragma unroll 4
    for (int c = 0; c < 192; ++c) {
      const int col = base + c;
      bf16 v;
      if (col < GNN_H) v = xs[col];
      else if (col < 2 * GNN_H) v = xr[col - GNN_H];
      else v = hp[col - 2 * GNN_H];
      sF[row * 384 + col] = v;
    }
  }
  __syncthreads();
  const bf16* A = sF + wave * 16 * 384;
  const int n = lane & 15, h = lane >> 4;
  for (int nt = 0; nt < 8; ++nt) {
    v8f c = gemm_tile<384>(A, 384, W1, 384, nt * 16, lane);
    const float bias = B1[nt * 16 + n];
#pragma unroll
    for (int r = 0; r < 8; ++r)
      sH[(wave * 16 + r + 8 * h) * GNN_H + nt * 16 + n] = (bf16)silu_f(c[r] + bias);
  }
  __syncthreads();
  msg_m2_scatter(sH + wave * 16 * GNN_H, W2, B2, recv, hbuf, sums, e0, wave, lane);
}

// Layer 1 edge MLP: edge_attr [E,24] (fp32) only, K padded 24->32.
__global__ __launch_bounds__(128) void edge_mlp_l1_kernel(
    const float* __restrict__ ea, const int* __restrict__ recv,
    const bf16* __restrict__ W1p /*[128,32] padded*/, const float* __restrict__ B1,
    const bf16* __restrict__ W2, const float* __restrict__ B2,
    bf16* __restrict__ hbuf, float* __restrict__ sums) {
  __shared__ bf16 sF[64 * 32];
  __shared__ bf16 sH[64 * 128];
  const int tid = threadIdx.x, lane = tid & 31, wave = tid >> 5;
  const long e0 = (long)blockIdx.x * 64;
  {
    const int row = tid >> 1, part = tid & 1;
    long e = e0 + row;
    if (e >= GNN_E) e = GNN_E - 1;
#pragma unroll
    for (int c = 0; c < 16; ++c) {
      const int col = part * 16 + c;
      sF[row * 32 + col] = (col < 24) ? (bf16)ea[e * 24 + col] : (bf16)0.0f;
    }
  }
  __syncthreads();
  const bf16* A = sF + wave * 16 * 32;
  const int n = lane & 15, h = lane >> 4;
  for (int nt = 0; nt < 8; ++nt) {
    v8f c = gemm_tile<32>(A, 32, W1p, 32, nt * 16, lane);
    const float bias = B1[nt * 16 + n];
#pragma unroll
    for (int r = 0; r < 8; ++r)
      sH[(wave * 16 + r + 8 * h) * GNN_H + nt * 16 + n] = (bf16)silu_f(c[r] + bias);
  }
  __syncthreads();
  msg_m2_scatter(sH + wave * 16 * GNN_H, W2, B2, recv, hbuf, sums, e0, wave, lane);
}

// Node update: x1 = res(x)/x + scatter-mean; x_out = x1 + u2(silu(u1(x1)))
template <int HAS_RES>
__global__ __launch_bounds__(128) void node_update_kernel(
    const float* __restrict__ xin8, float* __restrict__ xcur, bf16* __restrict__ xbf,
    const float* __restrict__ sums, const float* __restrict__ cnt,
    const float* __restrict__ Wres, const float* __restrict__ Bres,
    const bf16* __restrict__ Wu1, const float* __restrict__ Bu1,
    const bf16* __restrict__ Wu2, const float* __restrict__ Bu2) {
  __shared__ bf16 sX[64 * 128];  // 16 KB
  __shared__ bf16 sT[64 * 256];  // 32 KB
  const int tid = threadIdx.x, lane = tid & 31, wave = tid >> 5;
  const long i0 = (long)blockIdx.x * 64;
  {
    const int row = tid >> 1, part = tid & 1;
    long i = i0 + row;
    if (i >= GNN_N) i = GNN_N - 1;
    const float invc = 1.0f / fmaxf(cnt[i], 1.0f);
    float xl[8];
    if (HAS_RES) {
#pragma unroll
      for (int k = 0; k < 8; ++k) xl[k] = xin8[i * 8 + k];
    }
#pragma unroll 4
    for (int c = 0; c < 64; ++c) {
      const int col = part * 64 + c;
      float xr;
      if (HAS_RES) {
        xr = Bres[col];
#pragma unroll
        for (int k = 0; k < 8; ++k) xr += xl[k] * Wres[col * 8 + k];
      } else {
        xr = xcur[i * GNN_H + col];
      }
      const float x1 = xr + sums[i * GNN_H + col] * invc;
      sX[row * GNN_H + col] = (bf16)x1;
    }
  }
  __syncthreads();
  const int n = lane & 15, h = lane >> 4;
  const bf16* A = sX + wave * 16 * GNN_H;
  for (int nt = 0; nt < 16; ++nt) {  // u1: 128 -> 256
    v8f c = gemm_tile<128>(A, 128, Wu1, 128, nt * 16, lane);
    const float bias = Bu1[nt * 16 + n];
#pragma unroll
    for (int r = 0; r < 8; ++r)
      sT[(wave * 16 + r + 8 * h) * 256 + nt * 16 + n] = (bf16)silu_f(c[r] + bias);
  }
  __syncthreads();
  const bf16* A2 = sT + wave * 16 * 256;
  for (int nt = 0; nt < 8; ++nt) {  // u2: 256 -> 128 + residual
    v8f c = gemm_tile<256>(A2, 256, Wu2, 256, nt * 16, lane);
    const float bias = Bu2[nt * 16 + n];
    const int col = nt * 16 + n;
#pragma unroll
    for (int r = 0; r < 8; ++r) {
      const int m = wave * 16 + r + 8 * h;
      const long i = i0 + m;
      if (i < GNN_N) {
        const float v = (float)sX[m * GNN_H + col] + c[r] + bias;
        xcur[i * GNN_H + col] = v;
        xbf[i * GNN_H + col] = (bf16)v;
      }
    }
  }
}

__global__ __launch_bounds__(128) void out_mlp_kernel(
    const bf16* __restrict__ xbf, const bf16* __restrict__ Wo1,
    const float* __restrict__ Bo1, const bf16* __restrict__ Wo2,
    const float* __restrict__ Bo2, const bf16* __restrict__ Wo3p /*[16,128] padded*/,
    const float* __restrict__ Bo3, float* __restrict__ out) {
  __shared__ bf16 sA[64 * 128];
  __shared__ bf16 sT[64 * 128];
  const int tid = threadIdx.x, lane = tid & 31, wave = tid >> 5;
  const long i0 = (long)blockIdx.x * 64;
  {
    const int row = tid >> 1, part = tid & 1;
    long i = i0 + row;
    if (i >= GNN_N) i = GNN_N - 1;
#pragma unroll 8
    for (int c = 0; c < 64; ++c) {
      const int col = part * 64 + c;
      sA[row * 128 + col] = xbf[i * GNN_H + col];
    }
  }
  __syncthreads();
  const int n = lane & 15, h = lane >> 4;
  for (int nt = 0; nt < 8; ++nt) {
    v8f c = gemm_tile<128>(sA + wave * 16 * 128, 128, Wo1, 128, nt * 16, lane);
    const float bias = Bo1[nt * 16 + n];
#pragma unroll
    for (int r = 0; r < 8; ++r)
      sT[(wave * 16 + r + 8 * h) * 128 + nt * 16 + n] = (bf16)silu_f(c[r] + bias);
  }
  __syncthreads();
  for (int nt = 0; nt < 8; ++nt) {
    v8f c = gemm_tile<128>(sT + wave * 16 * 128, 128, Wo2, 128, nt * 16, lane);
    const float bias = Bo2[nt * 16 + n];
#pragma unroll
    for (int r = 0; r < 8; ++r)
      sA[(wave * 16 + r + 8 * h) * 128 + nt * 16 + n] = (bf16)silu_f(c[r] + bias);
  }
  __syncthreads();
  {
    v8f c = gemm_tile<128>(sA + wave * 16 * 128, 128, Wo3p, 128, 0, lane);
    if (n < 4) {
      const float bias = Bo3[n];
#pragma unroll
      for (int r = 0; r < 8; ++r) {
        const long i = i0 + wave * 16 + r + 8 * h;
        if (i < GNN_N) out[i * 4 + n] = c[r] + bias;
      }
    }
  }
}

__global__ void cvt_kernel(const float* __restrict__ src, bf16* __restrict__ dst,
                           int srows, int scols, int drows, int dcols) {
  const long total = (long)drows * dcols;
  for (long idx = (long)blockIdx.x * blockDim.x + threadIdx.x; idx < total;
       idx += (long)gridDim.x * blockDim.x) {
    const int r = (int)(idx / dcols), c = (int)(idx % dcols);
    const float v = (r < srows && c < scols) ? src[(long)r * scols + c] : 0.0f;
    dst[idx] = (bf16)v;
  }
}

__global__ void count_kernel(const int* __restrict__ recv, float* __restrict__ cnt) {
  const int e = blockIdx.x * 256 + threadIdx.x;
  if (e < GNN_E) atomicAdd(&cnt[recv[e]], 1.0f);
}

extern "C" void kernel_launch(void* const* d_in, const int* in_sizes, int n_in,
                              void* d_out, int out_size, void* d_ws, size_t ws_size,
                              hipStream_t stream) {
  (void)in_sizes; (void)out_size; (void)ws_size;
  const float* x8 = (const float*)d_in[0];
  const float* ea = (const float*)d_in[1];
  const int* edges = (const int*)d_in[2];
  const int* send = edges;
  const int* recv = edges + GNN_E;
  // Params flattened depth-first in dict order:
  // 3..12: l1 m1w,m1b,m2w,m2b,u1w,u1b,u2w,u2b,resw,resb
  // 13..20 / 21..28 / 29..36: l2/l3/l4 m1w,m1b,m2w,m2b,u1w,u1b,u2w,u2b
  // 37..42: o1w,o1b,o2w,o2b,o3w,o3b
  const float* P[43] = {};
  for (int i = 3; i < 43 && i < n_in; ++i) P[i] = (const float*)d_in[i];

  char* w = (char*)d_ws;
  auto carve = [&](size_t bytes) {
    void* r = (void*)w;
    w += (bytes + 255) & ~(size_t)255;
    return r;
  };
  bf16* xbf   = (bf16*)carve((size_t)GNN_N * GNN_H * sizeof(bf16));
  bf16* hbuf  = (bf16*)carve((size_t)GNN_E * GNN_H * sizeof(bf16));
  float* xcur = (float*)carve((size_t)GNN_N * GNN_H * sizeof(float));
  float* sums = (float*)carve((size_t)GNN_N * GNN_H * sizeof(float));
  float* cnt  = (float*)carve((size_t)GNN_N * sizeof(float));
  bf16* wb    = (bf16*)carve((size_t)600000 * sizeof(bf16));

  bf16* l1m1 = wb;
  bf16* l1m2 = l1m1 + 128 * 32;
  bf16* l1u1 = l1m2 + 128 * 128;
  bf16* l1u2 = l1u1 + 256 * 128;
  bf16 *lm1[3], *lm2[3], *lu1[3], *lu2[3];
  bf16* cur = l1u2 + 128 * 256;
  for (int l = 0; l < 3; ++l) {
    lm1[l] = cur; cur += 128 * 384;
    lm2[l] = cur; cur += 128 * 128;
    lu1[l] = cur; cur += 256 * 128;
    lu2[l] = cur; cur += 128 * 256;
  }
  bf16* o1 = cur; cur += 128 * 128;
  bf16* o2 = cur; cur += 128 * 128;
  bf16* o3 = cur;

  auto cvt = [&](const float* s, bf16* d, int sr, int sc, int dr, int dc) {
    long total = (long)dr * dc;
    int blocks = (int)((total + 255) / 256);
    if (blocks > 1024) blocks = 1024;
    cvt_kernel<<<blocks, 256, 0, stream>>>(s, d, sr, sc, dr, dc);
  };
  cvt(P[3], l1m1, 128, 24, 128, 32);
  cvt(P[5], l1m2, 128, 128, 128, 128);
  cvt(P[7], l1u1, 256, 128, 256, 128);
  cvt(P[9], l1u2, 128, 256, 128, 256);
  const int lb[3] = {13, 21, 29};
  for (int l = 0; l < 3; ++l) {
    cvt(P[lb[l] + 0], lm1[l], 128, 384, 128, 384);
    cvt(P[lb[l] + 2], lm2[l], 128, 128, 128, 128);
    cvt(P[lb[l] + 4], lu1[l], 256, 128, 256, 128);
    cvt(P[lb[l] + 6], lu2[l], 128, 256, 128, 256);
  }
  cvt(P[37], o1, 128, 128, 128, 128);
  cvt(P[39], o2, 128, 128, 128, 128);
  cvt(P[41], o3, 4, 128, 16, 128);

  hipMemsetAsync(cnt, 0, (size_t)GNN_N * sizeof(float), stream);
  count_kernel<<<(GNN_E + 255) / 256, 256, 0, stream>>>(recv, cnt);

  const int EB = GNN_E / 64;           // 6250 blocks of 64 edges
  const int NB = (GNN_N + 63) / 64;    // 782 blocks of 64 nodes

  // Layer 1
  hipMemsetAsync(sums, 0, (size_t)GNN_N * GNN_H * sizeof(float), stream);
  edge_mlp_l1_kernel<<<EB, 128, 0, stream>>>(ea, recv, l1m1, P[4], l1m2, P[6], hbuf,
                                             sums);
  node_update_kernel<1><<<NB, 128, 0, stream>>>(x8, xcur, xbf, sums, cnt, P[11], P[12],
                                                l1u1, P[8], l1u2, P[10]);
  // Layers 2-4
  for (int l = 0; l < 3; ++l) {
    hipMemsetAsync(sums, 0, (size_t)GNN_N * GNN_H * sizeof(float), stream);
    edge_mlp_kernel<<<EB, 128, 0, stream>>>(xbf, send, recv, lm1[l], P[lb[l] + 1],
                                            lm2[l], P[lb[l] + 3], hbuf, sums);
    node_update_kernel<0><<<NB, 128, 0, stream>>>(nullptr, xcur, xbf, sums, cnt,
                                                  nullptr, nullptr, lu1[l],
                                                  P[lb[l] + 5], lu2[l], P[lb[l] + 7]);
  }
  // Output MLP
  out_mlp_kernel<<<NB, 128, 0, stream>>>(xbf, o1, P[38], o2, P[40], o3, P[42],
                                         (float*)d_out);
}
